// Vgg16_31696858644836
// MI455X (gfx1250) — matrix operand order repaired and checked by reading, not compiled, not run on hardware
//
#include <hip/hip_runtime.h>
#include <cmath>

// ---------------------------------------------------------------------------
// Types / helpers
// ---------------------------------------------------------------------------
typedef __attribute__((ext_vector_type(16))) __bf16 bf16x16;
typedef __attribute__((ext_vector_type(8)))  float  v8f;

#define BATCH 4

__device__ __forceinline__ unsigned short f2bf(float f) {
    unsigned int u = __float_as_uint(f);
    unsigned int r = (u + 0x7FFFu + ((u >> 16) & 1u)) >> 16;   // RNE
    return (unsigned short)r;
}
__device__ __forceinline__ float bf2f(unsigned short h) {
    return __uint_as_float(((unsigned int)h) << 16);
}

// ---------------------------------------------------------------------------
// Weight repack: w[co][ci][3][3] f32  ->  wt16[tap][co][ci] bf16
// ---------------------------------------------------------------------------
__global__ __launch_bounds__(256) void k_wcvt(const float* __restrict__ w,
                                              unsigned short* __restrict__ dst,
                                              int Cout, int Cin) {
    int idx = blockIdx.x * 256 + threadIdx.x;
    int total = Cout * Cin * 9;
    if (idx >= total) return;
    int tap = idx % 9;
    int t2  = idx / 9;
    int ci  = t2 % Cin;
    int co  = t2 / Cin;
    dst[((size_t)(tap * Cout + co)) * Cin + ci] = f2bf(w[idx]);
}

// ---------------------------------------------------------------------------
// conv0: Cin=3 -> Cout=64, direct 3x3, f32 in, bf16 out, bias+relu
// ---------------------------------------------------------------------------
__global__ __launch_bounds__(256) void k_conv0(const float* __restrict__ X,
                                               const float* __restrict__ w,
                                               const float* __restrict__ b,
                                               unsigned short* __restrict__ out,
                                               int H, int W) {
    int idx = blockIdx.x * 256 + threadIdx.x;
    int total = BATCH * 64 * H * W;
    if (idx >= total) return;
    int wx = idx % W; int t = idx / W;
    int hy = t % H;   t /= H;
    int co = t % 64;  int bb = t / 64;
    float acc = b[co];
    #pragma unroll
    for (int ci = 0; ci < 3; ++ci) {
        #pragma unroll
        for (int ky = 0; ky < 3; ++ky) {
            int h = hy + ky - 1;
            if (h < 0 || h >= H) continue;
            #pragma unroll
            for (int kx = 0; kx < 3; ++kx) {
                int ww = wx + kx - 1;
                if (ww < 0 || ww >= W) continue;
                acc += X[((size_t)(bb * 3 + ci) * H + h) * W + ww] *
                       w[((co * 3 + ci) * 3 + ky) * 3 + kx];
            }
        }
    }
    out[idx] = f2bf(acc > 0.f ? acc : 0.f);
}

// ---------------------------------------------------------------------------
// WMMA implicit-GEMM conv: C[co][pix] = sum_k W[co][k] * im2col[k][pix]
// Block: 256 threads (8 waves). Tile: 64 co x 64 pixels. K chunk = 32 (bf16).
// Weight tile filled via GLOBAL_LOAD_ASYNC_TO_LDS_B128 (ASYNCcnt).
// ---------------------------------------------------------------------------
__global__ __launch_bounds__(256) void k_conv_wmma(
        const unsigned short* __restrict__ in,    // bf16 (B,Cin,H,W)
        const unsigned short* __restrict__ wt,    // bf16 (9,Cout,Cin)
        const float* __restrict__ bias,
        unsigned short* __restrict__ out,         // bf16 (B,Cout,H,W)
        int Cin, int Cout, int H, int W) {
    const int HW   = H * W;
    const int Ptot = BATCH * HW;

    __shared__ __align__(16) unsigned short Wlds[64][40];  // [co][ci]  rows = 80B
    __shared__ __align__(16) unsigned short Ilds[64][40];  // [pix][ci] rows = 80B

    const int t     = threadIdx.x;
    const int lane  = t & 31;
    const int wid   = t >> 5;
    const int waveM = wid >> 1;      // 0..3 : co groups of 16
    const int waveN = wid & 1;       // 0..1 : pixel groups of 32
    const int half  = lane >> 4;
    const int l16   = lane & 15;

    const int p0  = blockIdx.x * 64;   // pixel tile base (flattened b*H*W)
    const int co0 = blockIdx.y * 64;   // cout tile base

    v8f acc0 = {}; v8f acc1 = {};

    // ---- loader mappings ----
    // weights: one b128 async copy per thread
    const int w_col = t >> 2;          // 0..63 co_local
    const int w_ci0 = (t & 3) * 8;     // 0,8,16,24
    // im2col: one pixel per thread, 8 channels
    const int i_px  = t >> 2;          // 0..63 pixel_local
    const int i_ci0 = (t & 3) * 8;     // 0,8,16,24

    // decode this thread's pixel ONCE (fixed for whole kernel)
    const int p_ld  = p0 + i_px;
    const bool pv   = p_ld < Ptot;
    const int bb0   = p_ld / HW;
    const int prem  = p_ld - bb0 * HW;
    const int hh0   = prem / W;
    const int ww0   = prem - hh0 * W;
    const unsigned short* inb = in + (size_t)bb0 * Cin * HW;

    // LDS byte offsets (generic LDS pointer low 32 bits == LDS offset)
    const unsigned w_lds = (unsigned)(uintptr_t)&Wlds[w_col][w_ci0];

    const int nk = Cin >> 5;
    for (int tap = 0; tap < 9; ++tap) {
        const int dy = tap / 3 - 1;
        const int dx = tap % 3 - 1;
        const int hy = hh0 + dy, wx = ww0 + dx;
        const bool valid = pv && hy >= 0 && hy < H && wx >= 0 && wx < W;
        const int rowoff = hy * W + wx;
        for (int cc = 0; cc < nk; ++cc) {
            // ---- weight tile: async b128 global -> LDS ----
            {
                const unsigned short* src =
                    wt + ((size_t)(tap * Cout + co0 + w_col)) * Cin + cc * 32 + w_ci0;
                __builtin_prefetch(src + Cin, 0, 0);           // global_prefetch_b8
                asm volatile("global_load_async_to_lds_b128 %0, %1, off"
                             :: "v"(w_lds), "v"(src) : "memory");
            }
            // ---- im2col tile: 8 channels for this thread's pixel ----
            {
                const int cg = cc * 32 + i_ci0;
                unsigned short v[8];
                #pragma unroll
                for (int j = 0; j < 8; ++j)
                    v[j] = valid ? inb[(size_t)(cg + j) * HW + rowoff] : (unsigned short)0;
                uint4 pk;
                pk.x = (unsigned)v[0] | ((unsigned)v[1] << 16);
                pk.y = (unsigned)v[2] | ((unsigned)v[3] << 16);
                pk.z = (unsigned)v[4] | ((unsigned)v[5] << 16);
                pk.w = (unsigned)v[6] | ((unsigned)v[7] << 16);
                *(uint4*)&Ilds[i_px][i_ci0] = pk;
            }
            asm volatile("s_wait_asynccnt 0x0" ::: "memory");
            __syncthreads();

            // ---- assemble fragments (ISA 16-bit A/B lane layouts) ----
            union { bf16x16 v; unsigned int u[8]; } af, bfr0, bfr1;
            {
                const int mrow = waveM * 16 + l16;
                #pragma unroll
                for (int j = 0; j < 8; ++j) {
                    int kp = ((j < 4) ? (2 * j) : (2 * j + 8)) + 8 * half;
                    af.u[j] = *(const unsigned int*)&Wlds[mrow][kp];
                }
                const int n0 = waveN * 32 + l16;
                #pragma unroll
                for (int j = 0; j < 8; ++j) {
                    int kp = ((j < 4) ? (2 * j) : (2 * j + 8)) + 8 * half;
                    bfr0.u[j] = *(const unsigned int*)&Ilds[n0][kp];
                    bfr1.u[j] = *(const unsigned int*)&Ilds[n0 + 16][kp];
                }
            }
            acc0 = __builtin_amdgcn_wmma_f32_16x16x32_bf16(
                false, af.v, false, bfr0.v, (short)0, acc0, false, false);
            acc1 = __builtin_amdgcn_wmma_f32_16x16x32_bf16(
                false, af.v, false, bfr1.v, (short)0, acc1, false, false);
            __syncthreads();
        }
    }

    // ---- epilogue: bias + relu, bf16 NCHW stores ----
    const int p_a = p0 + waveN * 32 + l16;
    const int p_b = p_a + 16;
    const int ba = p_a / HW, ra = p_a - (p_a / HW) * HW;
    const int bb = p_b / HW, rb = p_b - (p_b / HW) * HW;
    #pragma unroll
    for (int r = 0; r < 8; ++r) {
        const int co = co0 + waveM * 16 + 8 * half + r;
        const float bv = bias[co];
        float v0 = acc0[r] + bv; v0 = v0 > 0.f ? v0 : 0.f;
        float v1 = acc1[r] + bv; v1 = v1 > 0.f ? v1 : 0.f;
        if (p_a < Ptot) out[((size_t)(ba * Cout + co)) * HW + ra] = f2bf(v0);
        if (p_b < Ptot) out[((size_t)(bb * Cout + co)) * HW + rb] = f2bf(v1);
    }
}

// ---------------------------------------------------------------------------
// 2x2 max pool, bf16 -> bf16
// ---------------------------------------------------------------------------
__global__ __launch_bounds__(256) void k_pool(const unsigned short* __restrict__ in,
                                              unsigned short* __restrict__ out,
                                              int C, int H, int W) {
    const int H2 = H >> 1, W2 = W >> 1;
    int idx = blockIdx.x * 256 + threadIdx.x;
    int total = BATCH * C * H2 * W2;
    if (idx >= total) return;
    int w2 = idx % W2; int t = idx / W2;
    int h2 = t % H2;   t /= H2;
    int c  = t % C;    int b = t / C;
    size_t base = ((size_t)(b * C + c) * H + 2 * h2) * W + 2 * w2;
    float m0 = bf2f(in[base]),       m1 = bf2f(in[base + 1]);
    float m2 = bf2f(in[base + W]),   m3 = bf2f(in[base + W + 1]);
    float m = m0 > m1 ? m0 : m1;
    m = m > m2 ? m : m2;
    m = m > m3 ? m : m3;
    out[idx] = f2bf(m);
}

// ---------------------------------------------------------------------------
// Hypercolumn bilinear gather -> flat (B, 960*64) f32, with batch reversal
// ---------------------------------------------------------------------------
__global__ __launch_bounds__(256) void k_gather(
        const unsigned short* __restrict__ o1, const unsigned short* __restrict__ o2,
        const unsigned short* __restrict__ o3, const unsigned short* __restrict__ o4,
        const int* __restrict__ ps, float* __restrict__ flat) {
    int idx = blockIdx.x * 256 + threadIdx.x;
    const int total = BATCH * 960 * 64;
    if (idx >= total) return;
    int s = idx & 63; int t = idx >> 6;
    int c = t % 960;  int b = t / 960;

    const unsigned short* buf; int C, Hl, cl; float comp, pad;
    if (c < 64)       { buf = o1; C = 64;  Hl = 224; cl = c;       comp = 1.f; pad = 0.5f; }
    else if (c < 192) { buf = o2; C = 128; Hl = 112; cl = c - 64;  comp = 2.f; pad = 1.5f; }
    else if (c < 448) { buf = o3; C = 256; Hl = 56;  cl = c - 192; comp = 4.f; pad = 3.5f; }
    else              { buf = o4; C = 512; Hl = 28;  cl = c - 448; comp = 8.f; pad = 7.5f; }
    const int Wl = Hl;

    float x = (float)ps[(b * 64 + s) * 3 + 1];
    float y = (float)ps[(b * 64 + s) * 3 + 2];
    float tx = (x - pad) / comp, ty = (y - pad) / comp;
    int fx = (int)floorf(tx), fy = (int)floorf(ty);
    int tx1 = fx     < 0 ? 0 : (fx     > Wl - 1 ? Wl - 1 : fx);
    int tx2 = fx + 1 < 0 ? 0 : (fx + 1 > Wl - 1 ? Wl - 1 : fx + 1);
    int ty1 = fy     < 0 ? 0 : (fy     > Hl - 1 ? Hl - 1 : fy);
    int ty2 = fy + 1 < 0 ? 0 : (fy + 1 > Hl - 1 ? Hl - 1 : fy + 1);

    size_t base = (size_t)(b * C + cl) * Hl * Wl;
    float Ia = bf2f(buf[base + (size_t)ty1 * Wl + tx1]);
    float Ib = bf2f(buf[base + (size_t)ty2 * Wl + tx1]);
    float Ic = bf2f(buf[base + (size_t)ty1 * Wl + tx2]);
    float Id = bf2f(buf[base + (size_t)ty2 * Wl + tx2]);

    float wa = ((float)tx2 - x) * ((float)ty2 - y);
    float wb = ((float)tx2 - x) * (y - (float)ty1);
    float wc = (x - (float)tx1) * ((float)ty2 - y);
    float wd = (x - (float)tx1) * (y - (float)ty1);

    flat[(size_t)(3 - b) * 61440 + c * 64 + s] = wa * Ia + wb * Ib + wc * Ic + wd * Id;
}

// ---------------------------------------------------------------------------
// MLP1: h1 = flat(4,61440) @ w1(61440,4096) + b1  (split-K, atomics, f32)
// ---------------------------------------------------------------------------
__global__ __launch_bounds__(256) void k_h1init(const float* __restrict__ b1,
                                                float* __restrict__ h1) {
    int idx = blockIdx.x * 256 + threadIdx.x;       // 16384
    if (idx < BATCH * 4096) h1[idx] = b1[idx & 4095];
}

__global__ __launch_bounds__(256) void k_mlp1(const float* __restrict__ flat,
                                              const float* __restrict__ w1,
                                              float* __restrict__ h1) {
    const int n  = blockIdx.x * 256 + threadIdx.x;  // 0..4095
    const int k0 = blockIdx.y * 3840;               // 16 chunks cover 61440
    float a0 = 0.f, a1 = 0.f, a2 = 0.f, a3 = 0.f;
    const float* f0 = flat;
    const float* f1 = flat + 61440;
    const float* f2 = flat + 2 * 61440;
    const float* f3 = flat + 3 * 61440;
    for (int k = k0; k < k0 + 3840; ++k) {
        float wv = w1[(size_t)k * 4096 + n];        // coalesced over n
        a0 += f0[k] * wv; a1 += f1[k] * wv;
        a2 += f2[k] * wv; a3 += f3[k] * wv;
    }
    atomicAdd(&h1[n],            a0);
    atomicAdd(&h1[4096 + n],     a1);
    atomicAdd(&h1[2 * 4096 + n], a2);
    atomicAdd(&h1[3 * 4096 + n], a3);
}

// ---------------------------------------------------------------------------
// MLP2: out = relu(h1) @ w2(4096,64) + b2   -> (4,64) f32
// ---------------------------------------------------------------------------
__global__ __launch_bounds__(256) void k_mlp2(const float* __restrict__ h1,
                                              const float* __restrict__ w2,
                                              const float* __restrict__ b2,
                                              float* __restrict__ out) {
    int t = threadIdx.x;          // 256 = 4*64
    int b = t >> 6, s = t & 63;
    float acc = 0.f;
    for (int n = 0; n < 4096; ++n) {
        float hv = h1[b * 4096 + n];
        hv = hv > 0.f ? hv : 0.f;
        acc += hv * w2[n * 64 + s];
    }
    out[b * 64 + s] = acc + b2[s];
}

// ---------------------------------------------------------------------------
// Host orchestration
// ---------------------------------------------------------------------------
static const int    CI[10]   = {3, 64, 64, 128, 128, 256, 256, 256, 512, 512};
static const int    CO[10]   = {64, 64, 128, 128, 256, 256, 256, 512, 512, 512};
static const size_t WOFF[10] = {0, 1728, 38592, 112320, 259776, 554688,
                                1144512, 1734336, 2913984, 5273280};

extern "C" void kernel_launch(void* const* d_in, const int* in_sizes, int n_in,
                              void* d_out, int out_size, void* d_ws, size_t ws_size,
                              hipStream_t stream) {
    (void)in_sizes; (void)n_in; (void)out_size; (void)ws_size;

    const float* X  = (const float*)d_in[0];
    const int*   ps = (const int*)d_in[1];
    const float* w[10]; const float* bi[10];
    for (int i = 0; i < 10; ++i) { w[i] = (const float*)d_in[2 + 2 * i]; bi[i] = (const float*)d_in[3 + 2 * i]; }
    const float* mw1 = (const float*)d_in[22];
    const float* mb1 = (const float*)d_in[23];
    const float* mw2 = (const float*)d_in[24];
    const float* mb2 = (const float*)d_in[25];
    float* out = (float*)d_out;

    // ---- carve workspace (bf16 regions then f32 regions) ----
    unsigned short* ws16 = (unsigned short*)d_ws;
    size_t o = 0;
    auto a16 = [&](size_t n) { unsigned short* p = ws16 + o; o += (n + 127) & ~(size_t)127; return p; };
    unsigned short* PING = a16(12845056);
    unsigned short* PONG = a16(12845056);
    unsigned short* O1   = a16(12845056);
    unsigned short* O2   = a16(6422528);
    unsigned short* O3   = a16(3211264);
    unsigned short* O4   = a16(1605632);
    unsigned short* WT   = a16(7632576);
    float* FLAT = (float*)(ws16 + o);
    float* H1   = FLAT + 245760;

    // ---- repack weights for layers 1..9 ----
    for (int i = 1; i < 10; ++i) {
        int total = CO[i] * CI[i] * 9;
        k_wcvt<<<(total + 255) / 256, 256, 0, stream>>>(w[i], WT + WOFF[i], CO[i], CI[i]);
    }

    // ---- conv stack ----
    k_conv0<<<(BATCH * 64 * 224 * 224 + 255) / 256, 256, 0, stream>>>(X, w[0], bi[0], PING, 224, 224);

    auto conv = [&](const unsigned short* src, int li, unsigned short* dst, int H, int W) {
        dim3 g((unsigned)((BATCH * H * W + 63) / 64), (unsigned)(CO[li] / 64));
        k_conv_wmma<<<g, 256, 0, stream>>>(src, WT + WOFF[li], bi[li], dst, CI[li], CO[li], H, W);
    };
    auto pool = [&](const unsigned short* src, unsigned short* dst, int C, int H, int W) {
        int total = BATCH * C * (H / 2) * (W / 2);
        k_pool<<<(total + 255) / 256, 256, 0, stream>>>(src, dst, C, H, W);
    };

    conv(PING, 1, O1, 224, 224);           // o1 tap (B,64,224,224)
    pool(O1, PING, 64, 224, 224);          // (B,64,112,112)
    conv(PING, 2, PONG, 112, 112);
    conv(PONG, 3, O2, 112, 112);           // o2 tap (B,128,112,112)
    pool(O2, PING, 128, 112, 112);
    conv(PING, 4, PONG, 56, 56);
    conv(PONG, 5, PING, 56, 56);
    conv(PING, 6, O3, 56, 56);             // o3 tap (B,256,56,56)
    pool(O3, PING, 256, 56, 56);
    conv(PING, 7, PONG, 28, 28);
    conv(PONG, 8, PING, 28, 28);
    conv(PING, 9, O4, 28, 28);             // o4 tap (B,512,28,28)

    // ---- hypercolumn gather ----
    k_gather<<<(BATCH * 960 * 64 + 255) / 256, 256, 0, stream>>>(O1, O2, O3, O4, ps, FLAT);

    // ---- MLP ----
    k_h1init<<<(BATCH * 4096 + 255) / 256, 256, 0, stream>>>(mb1, H1);
    dim3 g1(16, 16);                        // 4096 cols / 256  x  16 K-chunks
    k_mlp1<<<g1, 256, 0, stream>>>(FLAT, mw1, H1);
    k_mlp2<<<1, 256, 0, stream>>>(H1, mw2, mb2, out);
}
